// Anchor3DHead_61701500175350
// MI455X (gfx1250) — matrix operand order repaired and checked by reading, not compiled, not run on hardware
//
#include <hip/hip_runtime.h>

// Anchor3DHead 1x1-conv triple head as a single WMMA GEMM:
//   M = output channels (72, padded to 80 = 5 tiles of 16)
//   N = pixels (16 per wave-tile, contiguous in memory -> coalesced)
//   K = 384 input channels (12 steps of 32, f16 WMMA, f32 accumulate)

typedef __attribute__((ext_vector_type(16))) _Float16 v16h;
typedef __attribute__((ext_vector_type(8)))  float    v8f;

#define HW_      53568      // 248*216
#define C_IN     384
#define K_STEPS  12         // 384 / 32
#define M_TILES  5          // ceil(72/16)
#define A_FRAGS  (M_TILES * K_STEPS)        // 60 fragments of 1KB
#define A_HALVES (A_FRAGS * 32 * 16)        // 30720 f16 elements
#define A_BYTES  (A_HALVES * 2)             // 61440 bytes in d_ws
#define CLS_TOT  (72  * HW_)                // 4*18*HW
#define REG_TOT  (168 * HW_)                // 4*42*HW

__device__ __forceinline__ size_t out_idx(int b, int m, int pix) {
    if (m < 18) return (size_t)(b * 18 + m) * HW_ + pix;
    if (m < 60) return (size_t)CLS_TOT + (size_t)(b * 42 + (m - 18)) * HW_ + pix;
    return (size_t)CLS_TOT + (size_t)REG_TOT + (size_t)(b * 12 + (m - 60)) * HW_ + pix;
}

// Pack concatenated weights [K=384][M=72] into f16 WMMA A-fragment layout
// (contiguous k = 16*hi + i, not the ISA table order, 16-bit A 16x32): lane L in [0,16): M=L, halves cover
// K = {0..15}; lane L in [16,32): M=L-16, K = {16..31}.
__global__ __launch_bounds__(256) void prep_kernel(
    const float* __restrict__ w_cls, const float* __restrict__ b_cls,
    const float* __restrict__ w_reg, const float* __restrict__ b_reg,
    const float* __restrict__ w_dir, const float* __restrict__ b_dir,
    _Float16* __restrict__ a_out, float* __restrict__ bias_out)
{
    int idx = blockIdx.x * 256 + threadIdx.x;
    if (idx < A_HALVES) {
        int i    = idx & 15;          // half index within lane fragment
        int lane = (idx >> 4) & 31;
        int frag = idx >> 9;          // (mt*12 + ks)
        int ks   = frag % K_STEPS;
        int mt   = frag / K_STEPS;
        int hi   = lane >> 4;
        int v    = i >> 1, p = i & 1;
        int klocal = hi * 16 + v * 2 + p;
        int k = ks * 32 + klocal;
        int m = mt * 16 + (lane & 15);
        float val = 0.f;
        if (m < 72) {
            if      (m < 18) val = w_cls[k * 18 + m];
            else if (m < 60) val = w_reg[k * 42 + (m - 18)];
            else             val = w_dir[k * 12 + (m - 60)];
        }
        a_out[idx] = (_Float16)val;
    } else if (idx < A_HALVES + 72) {
        int j = idx - A_HALVES;
        bias_out[j] = (j < 18) ? b_cls[j] : (j < 60) ? b_reg[j - 18] : b_dir[j - 60];
    }
}

__global__ __launch_bounds__(256) void head_kernel(
    const float* __restrict__ x, const _Float16* __restrict__ aG,
    const float* __restrict__ bias, float* __restrict__ out)
{
    // 60KB of pre-swizzled A fragments, shared by all 8 waves of the block
    __shared__ v16h sA[A_FRAGS * 32];

    int tid = threadIdx.x;
    {   // cooperative 60KB L2->LDS copy (b128 per thread x 15)
        uint4*       s4 = (uint4*)sA;
        const uint4* g4 = (const uint4*)aG;
        #pragma unroll
        for (int t = 0; t < 15; ++t) s4[tid + t * 256] = g4[tid + t * 256];
    }
    __syncthreads();

    int wave = tid >> 5;
    int lane = tid & 31;
    int hi   = lane >> 4;       // lane group: selects K-half (B) / M-half (D)
    int col  = lane & 15;       // pixel within tile = WMMA N index

    int tile = blockIdx.x * 8 + wave;      // 13392 tiles total, exact
    int p0   = tile * 16;
    int b    = p0 / HW_;                   // 16 | HW so tiles never cross batch
    int pix  = (p0 - b * HW_) + col;

    // B fragment source: lane holds one pixel column, 16 consecutive channels
    // (offset by hi*16). Each global_load_b32 touches 2 contiguous 64B runs.
    const float* xb = x + (size_t)b * ((size_t)C_IN * HW_)
                        + (size_t)(hi * 16) * HW_ + pix;

    v8f acc[M_TILES];
    #pragma unroll
    for (int mt = 0; mt < M_TILES; ++mt) acc[mt] = (v8f){};

    for (int ks = 0; ks < K_STEPS; ++ks) {
        const float* xk = xb + (size_t)(ks * 32) * HW_;
        v16h bfrag;
        #pragma unroll
        for (int i = 0; i < 16; ++i) {
            float f = __builtin_nontemporal_load(xk + (size_t)i * HW_);
            bfrag[i] = (_Float16)f;        // v_cvt_f16_f32, f32 read once ever
        }
        #pragma unroll
        for (int mt = 0; mt < M_TILES; ++mt) {
            v16h af = sA[(mt * K_STEPS + ks) * 32 + lane];
            acc[mt] = __builtin_amdgcn_wmma_f32_16x16x32_f16(
                false, af, false, bfrag, (short)0, acc[mt], false, false);
        }
    }

    // D layout: VGPR r holds M = r (lanes 0-15) / M = r+8 (lanes 16-31), N = col
    int mb = hi * 8;
    #pragma unroll
    for (int mt = 0; mt < M_TILES; ++mt) {
        #pragma unroll
        for (int r = 0; r < 8; ++r) {
            int m = mt * 16 + mb + r;
            if (m < 72) {
                float vv = acc[mt][r] + bias[m];
                __builtin_nontemporal_store(vv, out + out_idx(b, m, pix));
            }
        }
    }
}

extern "C" void kernel_launch(void* const* d_in, const int* in_sizes, int n_in,
                              void* d_out, int out_size, void* d_ws, size_t ws_size,
                              hipStream_t stream) {
    (void)in_sizes; (void)n_in; (void)out_size; (void)ws_size;
    const float* x     = (const float*)d_in[0];
    const float* w_cls = (const float*)d_in[1];
    const float* b_cls = (const float*)d_in[2];
    const float* w_reg = (const float*)d_in[3];
    const float* b_reg = (const float*)d_in[4];
    const float* w_dir = (const float*)d_in[5];
    const float* b_dir = (const float*)d_in[6];

    _Float16* aG   = (_Float16*)d_ws;
    float*    bias = (float*)((char*)d_ws + A_BYTES);

    // pack weights into WMMA fragment layout + concat bias (tiny, runs once per launch)
    prep_kernel<<<(A_HALVES + 72 + 255) / 256, 256, 0, stream>>>(
        w_cls, b_cls, w_reg, b_reg, w_dir, b_dir, aG, bias);

    // 214272 pixels / 16 per wave / 8 waves per block = 1674 blocks (exact)
    head_kernel<<<1674, 256, 0, stream>>>(x, aG, bias, (float*)d_out);
}
